// edgesetMLP_6708738916903
// MI455X (gfx1250) — matrix operand order repaired and checked by reading, not compiled, not run on hardware
//
#include <hip/hip_runtime.h>
#include <hip/hip_bf16.h>

typedef _Float16 v16h __attribute__((ext_vector_type(16)));
typedef _Float16 v8h  __attribute__((ext_vector_type(8)));
typedef float    v8f  __attribute__((ext_vector_type(8)));

#define B_      16
#define NCGS    256
#define NATOMS  1024
#define KNN_    32
#define K_      128
#define DEPTH_  3
#define LDSS    136        // padded LDS row stride (halves): keeps 16B alignment, spreads banks

#define MODE_NONE 0
#define MODE_RELU 1
#define MODE_MEAN 2

// ---------------------------------------------------------------------------
// Pre-pass: convert 13 fp32 128x128 weight matrices to f16, transposed to
// N-major (wt[slot][n][k]) so WMMA B-fragments are contiguous 16B loads.
// slot(d,i) = d*4+i for {mlp_W1, mlp_W2, upd_W1, upd_W2}; slot 12 = dec_W1.
// ---------------------------------------------------------------------------
__global__ void convert_weights_kernel(const float* __restrict__ mlpW1,
                                       const float* __restrict__ mlpW2,
                                       const float* __restrict__ updW1,
                                       const float* __restrict__ updW2,
                                       const float* __restrict__ decW1,
                                       _Float16* __restrict__ wt)
{
    int e = blockIdx.x * blockDim.x + threadIdx.x;   // 0 .. 13*16384-1 (exact)
    int w = e >> 14;
    int r = e & 16383;
    int k = r >> 7;
    int n = r & 127;
    const float* src;
    if (w >= 12) {
        src = decW1;
    } else {
        int d = w >> 2, i = w & 3;
        const float* base = (i == 0) ? mlpW1 : (i == 1) ? mlpW2
                          : (i == 2) ? updW1 : updW2;
        src = base + d * (K_ * K_);
    }
    wt[w * (K_ * K_) + n * K_ + k] = (_Float16)src[k * K_ + n];
}

__global__ void zero_kernel(float* __restrict__ p, int n)
{
    int i = blockIdx.x * blockDim.x + threadIdx.x;
    if (i < n) p[i] = 0.0f;
}

// ---------------------------------------------------------------------------
// One (32x128) @ (128x128) GEMM tile, 4 waves, 16 WMMAs per wave.
// A: f16 LDS tile, row-major with stride LDSS (A-matrix 16x32 lane layout).
// B: f16 global, N-major (B-matrix 32x16 lane layout = two 8-half runs, 16 halves apart).
// Epilogue: +bias, optional relu, optional add-column-mean (cg_contract).
// ---------------------------------------------------------------------------
__device__ __forceinline__ void gemm32x128(const _Float16* __restrict__ src,
                                           _Float16* __restrict__ dst,
                                           const _Float16* __restrict__ wt,
                                           const float* __restrict__ bias,
                                           int mode, int tid)
{
    const int lane = tid & 31;
    const int wave = tid >> 5;
    const int half = lane >> 4;   // 0: lanes 0-15, 1: lanes 16-31
    const int l16  = lane & 15;

    v8f c[2][2] = {};             // [m-tile][n-local]

    #pragma unroll
    for (int kk = 0; kk < 4; ++kk) {           // K = 128 in steps of 32
        v16h A[2], Bm[2];
        #pragma unroll
        for (int m = 0; m < 2; ++m) {
            // 16-bit A 16x32: lanes<16 hold K {0..7,16..23}; lanes>=16 hold {8..15,24..31}
            const _Float16* p = src + (m * 16 + l16) * LDSS + kk * 32 + half * 8;
            v8h lo = *(const v8h*)p;
            v8h hi = *(const v8h*)(p + 16);
            #pragma unroll
            for (int i = 0; i < 8; ++i) { A[m][i] = lo[i]; A[m][i + 8] = hi[i]; }
        }
        #pragma unroll
        for (int nl = 0; nl < 2; ++nl) {
            // B 32x16: lane holds column l16, K {0..7,16..23} (half 0) / {8..15,24..31} (half 1), matching A
            int col = (wave * 2 + nl) * 16 + l16;
            const _Float16* q = wt + col * K_ + kk * 32 + half * 8;
            v8h blo = *(const v8h*)q;
            v8h bhi = *(const v8h*)(q + 16);
            #pragma unroll
            for (int i = 0; i < 8; ++i) { Bm[nl][i] = blo[i]; Bm[nl][i + 8] = bhi[i]; }
        }
        #pragma unroll
        for (int m = 0; m < 2; ++m)
            #pragma unroll
            for (int nl = 0; nl < 2; ++nl)
                c[m][nl] = __builtin_amdgcn_wmma_f32_16x16x32_f16(
                    false, A[m], false, Bm[nl], (short)0, c[m][nl], false, false);
    }

    #pragma unroll
    for (int nl = 0; nl < 2; ++nl) {
        int ntile = wave * 2 + nl;
        float bv = bias[ntile * 16 + l16];
        #pragma unroll
        for (int m = 0; m < 2; ++m)
            #pragma unroll
            for (int v = 0; v < 8; ++v)
                c[m][nl][v] += bv;

        if (mode == MODE_RELU) {
            #pragma unroll
            for (int m = 0; m < 2; ++m)
                #pragma unroll
                for (int v = 0; v < 8; ++v)
                    c[m][nl][v] = fmaxf(c[m][nl][v], 0.0f);
        } else if (mode == MODE_MEAN) {
            // cg_contract: mean over the 32 KNN rows of this column, added back.
            // Column lives in lanes L and L^16 (8 rows each per m-tile).
            float cs = 0.0f;
            #pragma unroll
            for (int m = 0; m < 2; ++m)
                #pragma unroll
                for (int v = 0; v < 8; ++v)
                    cs += c[m][nl][v];
            cs += __shfl_xor(cs, 16, 32);
            float mean = cs * (1.0f / (float)KNN_);
            #pragma unroll
            for (int m = 0; m < 2; ++m)
                #pragma unroll
                for (int v = 0; v < 8; ++v)
                    c[m][nl][v] += mean;
        }

        // C/D layout: VGPR v -> row v (lanes<16) or v+8 (lanes>=16), col = l16
        #pragma unroll
        for (int m = 0; m < 2; ++m)
            #pragma unroll
            for (int v = 0; v < 8; ++v)
                dst[(m * 16 + half * 8 + v) * LDSS + ntile * 16 + l16] =
                    (_Float16)c[m][nl][v];
    }
}

// ---------------------------------------------------------------------------
// Main fused kernel: one block per (batch, cg). RBF expand -> 13 GEMMs ->
// reduce Y against dist_vec into per-batch G[128][3] and S[3] (float atomics).
// ---------------------------------------------------------------------------
__global__ __launch_bounds__(128, 2)
void rbf_mlp_kernel(const float* __restrict__ cg_xyz,
                    const _Float16* __restrict__ wt,
                    const float* __restrict__ mlp_b1, const float* __restrict__ mlp_b2,
                    const float* __restrict__ upd_b1, const float* __restrict__ upd_b2,
                    const float* __restrict__ dec_b1,
                    float* __restrict__ G, float* __restrict__ S)
{
    __shared__ alignas(16) _Float16 bufA[KNN_ * LDSS];
    __shared__ alignas(16) _Float16 bufB[KNN_ * LDSS];
    __shared__ float s_dv[KNN_][3];
    __shared__ float s_dist[KNN_];

    const int tid = threadIdx.x;
    const int bx  = blockIdx.x;
    const int b   = bx >> 8;
    const int cgI = bx & 255;

    // Warm L2/L0 with the first weight matrix while the prolog runs.
    __builtin_prefetch(wt + tid * K_, 0, 0);

    if (tid < KNN_) {
        const float* cgb = cg_xyz + (size_t)b * NCGS * 3;
        float cx = cgb[cgI * 3 + 0], cy = cgb[cgI * 3 + 1], cz = cgb[cgI * 3 + 2];
        float nx = cgb[(1 + tid) * 3 + 0] - cx;
        float ny = cgb[(1 + tid) * 3 + 1] - cy;
        float nz = cgb[(1 + tid) * 3 + 2] - cz;
        s_dv[tid][0] = nx; s_dv[tid][1] = ny; s_dv[tid][2] = nz;
        s_dist[tid] = sqrtf(nx * nx + ny * ny + nz * nz);
    }
    __syncthreads();

    {   // RBF expansion: X[j][k] = exp(coeff*(dist[j]-offset[k])^2), thread = column k
        const float delta = 10.0f / 127.0f;
        const float coeff = -0.5f / (delta * delta);
        float off = (float)tid * delta;
        #pragma unroll 4
        for (int j = 0; j < KNN_; ++j) {
            float t = s_dist[j] - off;
            bufA[j * LDSS + tid] = (_Float16)__expf(coeff * t * t);
        }
    }
    __syncthreads();

    #pragma unroll 1
    for (int d = 0; d < DEPTH_; ++d) {
        gemm32x128(bufA, bufB, wt + (d * 4 + 0) * (K_ * K_), mlp_b1 + d * K_, MODE_RELU, tid);
        __syncthreads();
        gemm32x128(bufB, bufA, wt + (d * 4 + 1) * (K_ * K_), mlp_b2 + d * K_, MODE_MEAN, tid);
        __syncthreads();
        gemm32x128(bufA, bufB, wt + (d * 4 + 2) * (K_ * K_), upd_b1 + d * K_, MODE_RELU, tid);
        __syncthreads();
        gemm32x128(bufB, bufA, wt + (d * 4 + 3) * (K_ * K_), upd_b2 + d * K_, MODE_NONE, tid);
        __syncthreads();
    }
    gemm32x128(bufA, bufB, wt + 12 * (K_ * K_), dec_b1, MODE_RELU, tid);
    __syncthreads();

    // G[b][k][n] += sum_j Y[j][k] * dv[j][n]   (thread = column k)
    float g0 = 0.f, g1 = 0.f, g2 = 0.f;
    #pragma unroll 4
    for (int j = 0; j < KNN_; ++j) {
        float y = (float)bufB[j * LDSS + tid];
        g0 = fmaf(y, s_dv[j][0], g0);
        g1 = fmaf(y, s_dv[j][1], g1);
        g2 = fmaf(y, s_dv[j][2], g2);
    }
    float* gp = G + ((size_t)b * K_ + tid) * 3;
    atomicAdd(gp + 0, g0);
    atomicAdd(gp + 1, g1);
    atomicAdd(gp + 2, g2);

    if (tid < 3) {
        float s = 0.f;
        for (int j = 0; j < KNN_; ++j) s += s_dv[j][tid];
        atomicAdd(S + b * 3 + tid, s);
    }
}

// dx_recon[b,o,n] = sum_k dec_W2[k,o]*G[b,k,n] + dec_b2[o]*S[b,n]
__global__ void dx_recon_kernel(const float* __restrict__ decW2,
                                const float* __restrict__ decb2,
                                const float* __restrict__ G,
                                const float* __restrict__ S,
                                float* __restrict__ dx)
{
    int t = blockIdx.x * blockDim.x + threadIdx.x;   // b*1024 + o
    int b = t >> 10, o = t & 1023;
    float s0 = S[b * 3 + 0], s1 = S[b * 3 + 1], s2 = S[b * 3 + 2];
    float bb = decb2[o];
    float a0 = bb * s0, a1 = bb * s1, a2 = bb * s2;
    const float* gb = G + (size_t)b * K_ * 3;
    #pragma unroll 4
    for (int k = 0; k < K_; ++k) {
        float w = decW2[k * NATOMS + o];
        a0 = fmaf(w, gb[k * 3 + 0], a0);
        a1 = fmaf(w, gb[k * 3 + 1], a1);
        a2 = fmaf(w, gb[k * 3 + 2], a2);
    }
    dx[t * 3 + 0] = a0; dx[t * 3 + 1] = a1; dx[t * 3 + 2] = a2;
}

// cg_offset[b,j,n] = sum_i assign_norm[b,i,j] * dx_recon[b,i,n]
__global__ void cg_offset_kernel(const float* __restrict__ assign_norm,
                                 const float* __restrict__ dx,
                                 float* __restrict__ cgo)
{
    int b = blockIdx.x, j = threadIdx.x;             // 16 blocks x 256 threads
    const float* an  = assign_norm + (size_t)b * NATOMS * NCGS;
    const float* dxb = dx + (size_t)b * NATOMS * 3;
    float a0 = 0.f, a1 = 0.f, a2 = 0.f;
    #pragma unroll 4
    for (int i = 0; i < NATOMS; ++i) {
        float w = an[i * NCGS + j];
        a0 = fmaf(w, dxb[i * 3 + 0], a0);
        a1 = fmaf(w, dxb[i * 3 + 1], a1);
        a2 = fmaf(w, dxb[i * 3 + 2], a2);
    }
    float* o = cgo + ((size_t)b * NCGS + j) * 3;
    o[0] = a0; o[1] = a1; o[2] = a2;
}

__global__ void copy4_kernel(const float4* __restrict__ src, float4* __restrict__ dst)
{
    int i = blockIdx.x * blockDim.x + threadIdx.x;
    dst[i] = src[i];
}

// xyz_recon[b,a,:] = cg_xyz[b,idx[a],:] - cg_offset[b,idx[a],:] + dx_recon[b,a,:]
__global__ void recon_kernel(const float* __restrict__ cg_xyz,
                             const float* __restrict__ cgo,
                             const float* __restrict__ dx,
                             const int* __restrict__ assign_idx,
                             float* __restrict__ out)
{
    int t = blockIdx.x * blockDim.x + threadIdx.x;   // b*1024 + a
    int b = t >> 10, a = t & 1023;
    int j = assign_idx[a];
    #pragma unroll
    for (int n = 0; n < 3; ++n) {
        out[t * 3 + n] = cg_xyz[((size_t)b * NCGS + j) * 3 + n]
                       - cgo[((size_t)b * NCGS + j) * 3 + n]
                       + dx[t * 3 + n];
    }
}

extern "C" void kernel_launch(void* const* d_in, const int* in_sizes, int n_in,
                              void* d_out, int out_size, void* d_ws, size_t ws_size,
                              hipStream_t stream)
{
    const float* soft_assign = (const float*)d_in[0];
    const float* xyz         = (const float*)d_in[1];
    const float* cg_xyz      = (const float*)d_in[2];
    const float* assign_norm = (const float*)d_in[3];
    const int*   assign_idx  = (const int*)d_in[4];
    const float* mlp_W1 = (const float*)d_in[5];
    const float* mlp_b1 = (const float*)d_in[6];
    const float* mlp_W2 = (const float*)d_in[7];
    const float* mlp_b2 = (const float*)d_in[8];
    const float* upd_W1 = (const float*)d_in[9];
    const float* upd_b1 = (const float*)d_in[10];
    const float* upd_W2 = (const float*)d_in[11];
    const float* upd_b2 = (const float*)d_in[12];
    const float* dec_W1 = (const float*)d_in[13];
    const float* dec_b1 = (const float*)d_in[14];
    const float* dec_W2 = (const float*)d_in[15];
    const float* dec_b2 = (const float*)d_in[16];

    // Workspace layout (bytes):
    //   wt : 13*128*128 f16  = 425984
    //   G  : 16*128*3 f32    = 24576      (zeroed each call; contiguous with S)
    //   S  : 16*3 f32        = 192 (+pad to 256)
    //   dx : 16*1024*3 f32   = 196608
    //   cgo: 16*256*3 f32    = 49152      -> total ~680 KB
    char* ws = (char*)d_ws;
    _Float16* wt  = (_Float16*)ws;
    float*    G   = (float*)(ws + 425984);
    float*    S   = (float*)(ws + 425984 + 24576);
    float*    dx  = (float*)(ws + 425984 + 24576 + 256);
    float*    cgo = (float*)(ws + 425984 + 24576 + 256 + 196608);

    convert_weights_kernel<<<832, 256, 0, stream>>>(mlp_W1, mlp_W2, upd_W1, upd_W2,
                                                    dec_W1, wt);
    zero_kernel<<<25, 256, 0, stream>>>(G, B_ * K_ * 3 + B_ * 3);

    rbf_mlp_kernel<<<B_ * NCGS, 128, 0, stream>>>(cg_xyz, wt,
                                                  mlp_b1, mlp_b2, upd_b1, upd_b2,
                                                  dec_b1, G, S);

    dx_recon_kernel<<<(B_ * NATOMS) / 256, 256, 0, stream>>>(dec_W2, dec_b2, G, S, dx);
    cg_offset_kernel<<<B_, 256, 0, stream>>>(assign_norm, dx, cgo);

    float* out = (float*)d_out;
    const int soft_n = B_ * NATOMS * NCGS;           // 4194304
    const int xyz_n  = B_ * NATOMS * 3;              // 49152
    copy4_kernel<<<soft_n / 4 / 256, 256, 0, stream>>>((const float4*)soft_assign,
                                                       (float4*)out);
    copy4_kernel<<<xyz_n / 4 / 256, 256, 0, stream>>>((const float4*)xyz,
                                                      (float4*)(out + soft_n));
    recon_kernel<<<(B_ * NATOMS) / 256, 256, 0, stream>>>(cg_xyz, cgo, dx, assign_idx,
                                                          out + soft_n + xyz_n);
}
